// PointNetFeatureExtractPp_57492432224445
// MI455X (gfx1250) — matrix operand order repaired and checked by reading, not compiled, not run on hardware
//
#include <hip/hip_runtime.h>

// ---------------------------------------------------------------------------
// PointNet++ feature extraction for MI455X (gfx1250, wave32, WMMA 16x16x32 f16)
// ---------------------------------------------------------------------------

typedef __attribute__((ext_vector_type(16))) _Float16 v16h;
typedef __attribute__((ext_vector_type(8)))  float    v8f;

#define WMMA_F16(a, b, c) \
  __builtin_amdgcn_wmma_f32_16x16x32_f16(false, (a), false, (b), (short)0, (c), false, false)

#define NB 4
#define NM 8192
#define NS1 2048
#define NS2 512
#define NK 64

// ---- hardware tanh (V_TANH_F32 on gfx1250) ---------------------------------
#if __has_builtin(__builtin_amdgcn_tanhf)
static __device__ __forceinline__ float fast_tanh(float x) {
  return __builtin_amdgcn_tanhf(x);
}
#elif __has_builtin(__builtin_amdgcn_tanh_f32)
static __device__ __forceinline__ float fast_tanh(float x) {
  return __builtin_amdgcn_tanh_f32(x);
}
#else
static __device__ __forceinline__ float fast_tanh(float x) { return tanhf(x); }
#endif

// ---- A-fragment assembly from a row-major f16 LDS buffer -------------------
// 16-bit A 16x32 layout (ISA 7.12.2): lanes 0-15 hold K=0..7 (vgpr0-3) and
// K=16..23 (vgpr4-7); lanes 16-31 hold K=8..15 and K=24..31. m = lane&15.
static __device__ __forceinline__ v16h a_frag_lds(const _Float16* base, int ld, int kbase) {
  const int l = threadIdx.x & 31;
  const int m = l & 15;
  const int g = l >> 4;
  v16h a;
#pragma unroll
  for (int e = 0; e < 16; ++e) {
    const int v = e >> 1;
    const int k = ((v & 4) ? 16 : 0) + ((v & 3) << 1) + (e & 1) + (g << 3);
    a[e] = base[m * ld + kbase + k];
  }
  return a;
}

// ---- B-fragment load from pre-packed weights -------------------------------
// pack layout: fragment f, lane l -> 16 contiguous halves at (f*32 + l)*16.
static __device__ __forceinline__ v16h b_frag(const _Float16* pw, int frag) {
  const int l = threadIdx.x & 31;
  return *(const v16h*)(pw + (size_t)(frag * 32 + l) * 16);
}

// ---------------------------------------------------------------------------
// Weight packing: W [fin, fout] fp32 -> f16 WMMA B-fragment tiles, zero padded
// B 32x16 layout: lane l holds column n = l&15; half e -> k = (e&7) + 16*(e>>3) + 8*(l>>4).
// ---------------------------------------------------------------------------
__global__ void pack_w(const float* __restrict__ W, _Float16* __restrict__ out,
                       int fin, int fout, int Kp, int Np) {
  const int NT = Np >> 4;
  const int KT = Kp >> 5;
  const int total = KT * NT * 512;
  for (int t = blockIdx.x * blockDim.x + threadIdx.x; t < total;
       t += gridDim.x * blockDim.x) {
    const int e = t & 15;
    const int l = (t >> 4) & 31;
    const int f = t >> 9;
    const int nt = f % NT;
    const int kt = f / NT;
    const int k = kt * 32 + (e & 7) + ((e >> 3) << 4) + ((l >> 4) << 3);
    const int n = nt * 16 + (l & 15);
    float v = (k < fin && n < fout) ? W[(size_t)k * fout + n] : 0.0f;
    out[t] = (_Float16)v;
  }
}

// ---------------------------------------------------------------------------
// Farthest point sampling: one block per batch. pos [Mi,3] -> centers [n,3]
// ---------------------------------------------------------------------------
__global__ void __launch_bounds__(256) fps_kernel(const float* __restrict__ pos,
                                                  float* __restrict__ centers,
                                                  int Mi, int n) {
  __shared__ float dist[NM];
  __shared__ float wd[8];
  __shared__ int   wi[8];
  __shared__ float bpt[3];

  const int tid = threadIdx.x;
  const float* P = pos + (size_t)blockIdx.x * Mi * 3;
  float* C = centers + (size_t)blockIdx.x * n * 3;

  const float x0 = P[0], y0 = P[1], z0 = P[2];
  for (int j = tid; j < Mi; j += 256) {
    float dx = P[j * 3] - x0, dy = P[j * 3 + 1] - y0, dz = P[j * 3 + 2] - z0;
    dist[j] = dx * dx + dy * dy + dz * dz;
  }
  if (tid == 0) { C[0] = x0; C[1] = y0; C[2] = z0; }
  __syncthreads();

  for (int i = 1; i < n; ++i) {
    float bd = -1.0f;
    int bi = 0;
    for (int j = tid; j < Mi; j += 256) {
      float d = dist[j];
      if (d > bd) { bd = d; bi = j; }
    }
#pragma unroll
    for (int off = 16; off > 0; off >>= 1) {
      float od = __shfl_down(bd, off, 32);
      int   oi = __shfl_down(bi, off, 32);
      if (od > bd) { bd = od; bi = oi; }
    }
    if ((tid & 31) == 0) { wd[tid >> 5] = bd; wi[tid >> 5] = bi; }
    __syncthreads();
    if (tid == 0) {
      for (int w = 1; w < 8; ++w)
        if (wd[w] > bd) { bd = wd[w]; bi = wi[w]; }
      float bx = P[bi * 3], by = P[bi * 3 + 1], bz = P[bi * 3 + 2];
      C[i * 3] = bx; C[i * 3 + 1] = by; C[i * 3 + 2] = bz;
      bpt[0] = bx; bpt[1] = by; bpt[2] = bz;
    }
    __syncthreads();
    const float bx = bpt[0], by = bpt[1], bz = bpt[2];
    for (int j = tid; j < Mi; j += 256) {
      float dx = P[j * 3] - bx, dy = P[j * 3 + 1] - by, dz = P[j * 3 + 2] - bz;
      dist[j] = fminf(dist[j], dx * dx + dy * dy + dz * dz);
    }
    __syncthreads();
  }
}

// ---------------------------------------------------------------------------
// Local branch: tanh-MLP 3 -> 64 -> 64 -> 128 over all B*M points.
// Block = 128 threads = 4 waves; each wave owns a 16-point row tile.
// ---------------------------------------------------------------------------
__global__ void __launch_bounds__(128) local_mlp_kernel(
    const float* __restrict__ pos,
    const _Float16* __restrict__ pw0, const float* __restrict__ b0,
    const _Float16* __restrict__ pw1, const float* __restrict__ b1,
    const _Float16* __restrict__ pw2, const float* __restrict__ b2,
    float* __restrict__ out) {
  __shared__ _Float16 h1buf[4][16 * 64];
  __shared__ _Float16 h2buf[4][16 * 64];

  const int tid = threadIdx.x, wid = tid >> 5, lane = tid & 31;
  const int g = lane >> 4, nl = lane & 15;
  const int p0 = blockIdx.x * 64 + wid * 16;

  __builtin_prefetch(pos + (size_t)(p0 + 64) * 3, 0, 1);

  // layer 1: A tile from global positions (K padded 3 -> 32 with zeros)
  v16h a0;
#pragma unroll
  for (int e = 0; e < 16; ++e) {
    const int v = e >> 1;
    const int k = ((v & 4) ? 16 : 0) + ((v & 3) << 1) + (e & 1) + (g << 3);
    float val = 0.0f;
    if (k < 3) val = pos[(size_t)(p0 + nl) * 3 + k];
    a0[e] = (_Float16)val;
  }
#pragma unroll
  for (int nt = 0; nt < 4; ++nt) {
    v8f acc = {};
    acc = WMMA_F16(a0, b_frag(pw0, nt), acc);
    const float bias = b0[nt * 16 + nl];
#pragma unroll
    for (int r = 0; r < 8; ++r)
      h1buf[wid][(r + 8 * g) * 64 + nt * 16 + nl] = (_Float16)fast_tanh(acc[r] + bias);
  }
  __syncthreads();

  // layer 2: 64 -> 64  (A fragments hoisted into registers, reused over nt)
  {
    v16h af[2];
#pragma unroll
    for (int kt = 0; kt < 2; ++kt) af[kt] = a_frag_lds(h1buf[wid], 64, kt * 32);
#pragma unroll
    for (int nt = 0; nt < 4; ++nt) {
      v8f acc = {};
#pragma unroll
      for (int kt = 0; kt < 2; ++kt)
        acc = WMMA_F16(af[kt], b_frag(pw1, kt * 4 + nt), acc);
      const float bias = b1[nt * 16 + nl];
#pragma unroll
      for (int r = 0; r < 8; ++r)
        h2buf[wid][(r + 8 * g) * 64 + nt * 16 + nl] = (_Float16)fast_tanh(acc[r] + bias);
    }
  }
  __syncthreads();

  // layer 3: 64 -> 128, store fp32 to d_out
  {
    v16h af[2];
#pragma unroll
    for (int kt = 0; kt < 2; ++kt) af[kt] = a_frag_lds(h2buf[wid], 64, kt * 32);
#pragma unroll
    for (int nt = 0; nt < 8; ++nt) {
      v8f acc = {};
#pragma unroll
      for (int kt = 0; kt < 2; ++kt)
        acc = WMMA_F16(af[kt], b_frag(pw2, kt * 8 + nt), acc);
      const float bias = b2[nt * 16 + nl];
#pragma unroll
      for (int r = 0; r < 8; ++r)
        out[(size_t)(p0 + r + 8 * g) * 128 + nt * 16 + nl] = fast_tanh(acc[r] + bias);
    }
  }
}

// ---------------------------------------------------------------------------
// Set-abstraction layer: one block (4 waves) per group.
//   phase 1: wave 0 ball query (first K=64 in-radius points) via ballot
//   phase 2: cooperative gather of group features into LDS (f16, K-padded)
//   phase 3: per-wave row-tile WMMA MLP (CINP->CHID->COUT, tanh)
//   phase 4: masked max pool over valid rows -> hout[b][s][COUT]
// MODE 1: feats = [pos | x | rel] (9 ch, pad 32). MODE 2: [h1 | rel] (131, pad 160)
// ---------------------------------------------------------------------------
template <int MODE, int S, int MI, int CINP, int CHID, int COUT>
__global__ void __launch_bounds__(128) sa_kernel(
    const float* __restrict__ pos, const float* __restrict__ xin,
    const float* __restrict__ centers,
    const _Float16* __restrict__ pw1, const float* __restrict__ b1,
    const _Float16* __restrict__ pw2, const float* __restrict__ b2,
    float* __restrict__ hout, float r2) {
  constexpr int KT1 = CINP / 32;
  constexpr int NT1 = CHID / 16;
  constexpr int KT2 = CHID / 32;
  constexpr int NT2 = COUT / 16;
  constexpr int XS = (MODE == 1) ? 3 : 128;

  __shared__ _Float16 feats[64 * CINP];
  __shared__ _Float16 hbuf[64 * CHID];
  __shared__ int nidx[NK];
  __shared__ int scount;
  __shared__ float pool[4 * COUT];

  const int tid = threadIdx.x, wid = tid >> 5, lane = tid & 31;
  const int s = blockIdx.x, b = blockIdx.y;
  const float* Pb = pos + (size_t)b * MI * 3;
  const float* Xb = xin + (size_t)b * MI * XS;
  const float* C = centers + ((size_t)b * S + s) * 3;
  const float cx = C[0], cy = C[1], cz = C[2];

  // -------- ball query (wave 0) --------
  if (wid == 0) {
    int cnt = 0;
    for (int base = 0; base < MI; base += 32) {
      const int j = base + lane;
      float d2v = 1e30f;
      if (j < MI) {
        float dx = Pb[j * 3] - cx, dy = Pb[j * 3 + 1] - cy, dz = Pb[j * 3 + 2] - cz;
        d2v = dx * dx + dy * dy + dz * dz;
      }
      const bool inb = d2v <= r2;
      const unsigned mlo = (unsigned)__ballot(inb);
      const int prefix = __popc(mlo & ((1u << lane) - 1u));
      if (inb && (cnt + prefix) < NK) nidx[cnt + prefix] = j;
      cnt += __popc(mlo);
      if (cnt >= NK) break;
    }
    if (lane == 0) scount = (cnt > NK) ? NK : cnt;
  }
  __syncthreads();
  const int count = scount;

  // -------- gather features into LDS (row-major [64][CINP] f16) --------
  for (int t = tid; t < 64 * CINP; t += 128) {
    const int row = t / CINP;
    const int k = t - row * CINP;
    float v = 0.0f;
    if (row < count) {
      const int idx = nidx[row];
      if (MODE == 1) {
        if (k < 3)       v = Pb[idx * 3 + k];
        else if (k < 6)  v = Xb[idx * 3 + (k - 3)];
        else if (k == 6) v = Pb[idx * 3 + 0] - cx;
        else if (k == 7) v = Pb[idx * 3 + 1] - cy;
        else if (k == 8) v = Pb[idx * 3 + 2] - cz;
      } else {
        if (k < 128)       v = Xb[(size_t)idx * 128 + k];
        else if (k == 128) v = Pb[idx * 3 + 0] - cx;
        else if (k == 129) v = Pb[idx * 3 + 1] - cy;
        else if (k == 130) v = Pb[idx * 3 + 2] - cz;
      }
    }
    feats[t] = (_Float16)v;
  }
  __syncthreads();

  const int g = lane >> 4, nl = lane & 15;
  const int rt = wid;  // row tile 0..3 (rows rt*16 .. rt*16+15 of the group)

  // -------- layer 1: CINP -> CHID --------
  {
    v16h af[KT1];
#pragma unroll
    for (int kt = 0; kt < KT1; ++kt)
      af[kt] = a_frag_lds(feats + rt * 16 * CINP, CINP, kt * 32);
    for (int nt = 0; nt < NT1; ++nt) {
      v8f acc = {};
#pragma unroll
      for (int kt = 0; kt < KT1; ++kt)
        acc = WMMA_F16(af[kt], b_frag(pw1, kt * NT1 + nt), acc);
      const float bias = b1[nt * 16 + nl];
#pragma unroll
      for (int r = 0; r < 8; ++r)
        hbuf[(rt * 16 + r + 8 * g) * CHID + nt * 16 + nl] = (_Float16)fast_tanh(acc[r] + bias);
    }
  }
  __syncthreads();

  // -------- layer 2: CHID -> COUT, masked max pool --------
  {
    v16h af[KT2];
#pragma unroll
    for (int kt = 0; kt < KT2; ++kt)
      af[kt] = a_frag_lds(hbuf + rt * 16 * CHID, CHID, kt * 32);
    for (int nt = 0; nt < NT2; ++nt) {
      v8f acc = {};
#pragma unroll
      for (int kt = 0; kt < KT2; ++kt)
        acc = WMMA_F16(af[kt], b_frag(pw2, kt * NT2 + nt), acc);
      const float bias = b2[nt * 16 + nl];
      float vmax = -3.0e38f;
#pragma unroll
      for (int r = 0; r < 8; ++r) {
        const int row = rt * 16 + r + 8 * g;
        const float hv = fast_tanh(acc[r] + bias);
        if (row < count) vmax = fmaxf(vmax, hv);
      }
      vmax = fmaxf(vmax, __shfl_xor(vmax, 16, 32));
      if (g == 0) pool[wid * COUT + nt * 16 + nl] = vmax;
    }
  }
  __syncthreads();

  for (int c = tid; c < COUT; c += 128) {
    const float m0 = fmaxf(pool[c], pool[COUT + c]);
    const float m1 = fmaxf(pool[2 * COUT + c], pool[3 * COUT + c]);
    hout[((size_t)b * S + s) * COUT + c] = fmaxf(m0, m1);
  }
}

// ---------------------------------------------------------------------------
// Final global feature: max over 512 groups of h2 [B,512,256] -> [B,256]
// ---------------------------------------------------------------------------
__global__ void __launch_bounds__(256) gmax_kernel(const float* __restrict__ h2,
                                                   float* __restrict__ out) {
  const int b = blockIdx.x, c = threadIdx.x;
  float m = -3.0e38f;
  for (int s = 0; s < NS2; ++s) m = fmaxf(m, h2[((size_t)b * NS2 + s) * 256 + c]);
  out[b * 256 + c] = m;
}

// ---------------------------------------------------------------------------
extern "C" void kernel_launch(void* const* d_in, const int* in_sizes, int n_in,
                              void* d_out, int out_size, void* d_ws, size_t ws_size,
                              hipStream_t stream) {
  (void)in_sizes; (void)n_in; (void)out_size; (void)ws_size;

  const float* x    = (const float*)d_in[0];
  const float* pos  = (const float*)d_in[1];
  const float* lw0  = (const float*)d_in[2];
  const float* lb0  = (const float*)d_in[3];
  const float* lw1  = (const float*)d_in[4];
  const float* lb1  = (const float*)d_in[5];
  const float* lw2  = (const float*)d_in[6];
  const float* lb2  = (const float*)d_in[7];
  const float* g1w0 = (const float*)d_in[8];
  const float* g1b0 = (const float*)d_in[9];
  const float* g1w1 = (const float*)d_in[10];
  const float* g1b1 = (const float*)d_in[11];
  const float* g2w0 = (const float*)d_in[12];
  const float* g2b0 = (const float*)d_in[13];
  const float* g2w1 = (const float*)d_in[14];
  const float* g2b1 = (const float*)d_in[15];
  float* out = (float*)d_out;

  // --- carve workspace ---
  char* ws = (char*)d_ws;
  size_t off = 0;
  auto alloc = [&](size_t bytes) -> void* {
    off = (off + 255) & ~(size_t)255;
    void* p = ws + off;
    off += bytes;
    return p;
  };
  _Float16* pw_l0  = (_Float16*)alloc((size_t)1 * 4  * 512 * 2);  // 32x64
  _Float16* pw_l1  = (_Float16*)alloc((size_t)2 * 4  * 512 * 2);  // 64x64
  _Float16* pw_l2  = (_Float16*)alloc((size_t)2 * 8  * 512 * 2);  // 64x128
  _Float16* pw_g10 = (_Float16*)alloc((size_t)1 * 4  * 512 * 2);  // 32x64
  _Float16* pw_g11 = (_Float16*)alloc((size_t)2 * 8  * 512 * 2);  // 64x128
  _Float16* pw_g20 = (_Float16*)alloc((size_t)5 * 8  * 512 * 2);  // 160x128
  _Float16* pw_g21 = (_Float16*)alloc((size_t)4 * 16 * 512 * 2);  // 128x256
  float* c1 = (float*)alloc((size_t)NB * NS1 * 3 * 4);
  float* h1 = (float*)alloc((size_t)NB * NS1 * 128 * 4);
  float* c2 = (float*)alloc((size_t)NB * NS2 * 3 * 4);
  float* h2 = (float*)alloc((size_t)NB * NS2 * 256 * 4);

  // --- 1. pack weights into WMMA B-fragment layout ---
  pack_w<<<64, 256, 0, stream>>>(lw0,  pw_l0,  3,   64,  32,  64);
  pack_w<<<64, 256, 0, stream>>>(lw1,  pw_l1,  64,  64,  64,  64);
  pack_w<<<64, 256, 0, stream>>>(lw2,  pw_l2,  64,  128, 64,  128);
  pack_w<<<64, 256, 0, stream>>>(g1w0, pw_g10, 9,   64,  32,  64);
  pack_w<<<64, 256, 0, stream>>>(g1w1, pw_g11, 64,  128, 64,  128);
  pack_w<<<64, 256, 0, stream>>>(g2w0, pw_g20, 131, 128, 160, 128);
  pack_w<<<64, 256, 0, stream>>>(g2w1, pw_g21, 128, 256, 128, 256);

  // --- 2. local branch MLP (independent of SA path) ---
  local_mlp_kernel<<<(NB * NM) / 64, 128, 0, stream>>>(
      pos, pw_l0, lb0, pw_l1, lb1, pw_l2, lb2, out);

  // --- 3. FPS 8192 -> 2048 ---
  fps_kernel<<<NB, 256, 0, stream>>>(pos, c1, NM, NS1);

  // --- 4. SA1: ball query r=0.2, MLP 9->64->128, pool -> h1 [B,2048,128] ---
  sa_kernel<1, NS1, NM, 32, 64, 128><<<dim3(NS1, NB), 128, 0, stream>>>(
      pos, x, c1, pw_g10, g1b0, pw_g11, g1b1, h1, 0.04f);

  // --- 5. FPS 2048 -> 512 on SA1 centers ---
  fps_kernel<<<NB, 256, 0, stream>>>(c1, c2, NS1, NS2);

  // --- 6. SA2: ball query r=0.4, MLP 131->128->256, pool -> h2 [B,512,256] ---
  sa_kernel<2, NS2, NS1, 160, 128, 256><<<dim3(NS2, NB), 128, 0, stream>>>(
      c1, h1, c2, pw_g20, g2b0, pw_g21, g2b1, h2, 0.16f);

  // --- 7. global max over groups -> tail of d_out ---
  gmax_kernel<<<NB, 256, 0, stream>>>(h2, out + (size_t)NB * NM * 128);
}
